// MK_MMD_Loss_28630251995842
// MI455X (gfx1250) — compile-verified, matrix operand order checked by reading, not run on hardware
//
#include <hip/hip_runtime.h>
#include <hip/hip_bf16.h>
#include <math.h>

// ---------------------------------------------------------------------------
// MK-MMD loss for MI455X (gfx1250), wave32, WMMA core.
//   source: 2048x512 f32, target: 2048x512 f32, beta: 5 f32 -> scalar f32
// ---------------------------------------------------------------------------

#define NS    2048
#define N_TOT 4096
#define DIM   512
#define KNUM  5
#define EPSQ  0.001f
#define PPAIR 1024

typedef __attribute__((ext_vector_type(2)))  float v2f;
typedef __attribute__((ext_vector_type(8)))  float v8f;

#if defined(__has_builtin)
#if __has_builtin(__builtin_amdgcn_wmma_f32_16x16x4_f32)
#define HAVE_WMMA_F32X4 1
#endif
#endif

#ifndef HAVE_WMMA_F32X4
typedef __attribute__((ext_vector_type(16))) __bf16 v16bf;
#endif

// ---------------- workspace layout (bytes) ----------------
static constexpr size_t OFF_DMAT  = 0;                                          // 4096*4096 f32 = 64MB
static constexpr size_t OFF_SQ    = OFF_DMAT  + (size_t)N_TOT * N_TOT * 4;      // 4096 f32
static constexpr size_t OFF_TOTAL = OFF_SQ    + (size_t)N_TOT * 4;              // 4096*512 f32 = 8MB
static constexpr size_t OFF_G     = OFF_TOTAL + (size_t)N_TOT * DIM * 4;        // 1024*5 f32
static constexpr size_t OFF_BETA  = OFF_G     + (size_t)PPAIR * KNUM * 4;       // 5 f32 (padded)
static constexpr size_t OFF_STATE = OFF_BETA  + 32;                             // lo,hi f32 ; rank u64
static constexpr size_t OFF_MAX   = OFF_STATE + 16;                             // int
static constexpr size_t OFF_HIST  = OFF_MAX   + 16;                             // 512 u32
static constexpr size_t OFF_SUMS  = OFF_HIST  + 512 * 4;                        // 3 f64

// ---------------- small helpers ----------------
__global__ __launch_bounds__(512) void zero_kernel(unsigned* hist, double* sums, int* maxbits) {
  int t = threadIdx.x;
  if (t < 512) hist[t] = 0u;
  if (t < 3)   sums[t] = 0.0;
  if (t == 0)  *maxbits = 0;
}

__global__ __launch_bounds__(256) void concat_kernel(const float* __restrict__ src,
                                                     const float* __restrict__ tgt,
                                                     float* __restrict__ total) {
  int n = N_TOT * DIM;
  for (int i = blockIdx.x * blockDim.x + threadIdx.x; i < n; i += gridDim.x * blockDim.x)
    total[i] = (i < NS * DIM) ? src[i] : tgt[i - NS * DIM];
}

__global__ __launch_bounds__(128) void sqnorm_kernel(const float* __restrict__ T,
                                                     float* __restrict__ sq) {
  __shared__ float red[128];
  int row = blockIdx.x, t = threadIdx.x;
  const float* r = T + (size_t)row * DIM;
  float s = 0.0f;
  for (int j = t; j < DIM; j += 128) { float v = r[j]; s += v * v; }
  red[t] = s; __syncthreads();
  for (int off = 64; off > 0; off >>= 1) {
    if (t < off) red[t] += red[t + off];
    __syncthreads();
  }
  if (t == 0) sq[row] = red[0];
}

// ---------------- WMMA GEMM + squared distances ----------------
// block = 256 threads (8 waves); block tile 128x128; wave tile 32x64
// D[i][j] = max(0, sq[i] + sq[j] - 2 * dot(total_i, total_j))
__global__ __launch_bounds__(256) void gemm_dist_kernel(const float* __restrict__ T,
                                                        const float* __restrict__ sq,
                                                        float* __restrict__ Dm,
                                                        int* __restrict__ maxbits) {
  const int lane = threadIdx.x & 31;
  const int w    = threadIdx.x >> 5;
  const int wrow = w & 3;          // 0..3  -> 32-row slab
  const int wcol = w >> 2;         // 0..1  -> 64-col slab
  const int mbase = blockIdx.x * 128 + wrow * 32;
  const int nbase = blockIdx.y * 128 + wcol * 64;
  const int l15 = lane & 15;
  const int h   = lane >> 4;       // 0 or 1 (half-wave)

  const float* arow[2];
  const float* brow[4];
#pragma unroll
  for (int mi = 0; mi < 2; ++mi) arow[mi] = T + (size_t)(mbase + mi * 16 + l15) * DIM;
#pragma unroll
  for (int ni = 0; ni < 4; ++ni) brow[ni] = T + (size_t)(nbase + ni * 16 + l15) * DIM;

  v8f acc[2][4] = {};

#ifdef HAVE_WMMA_F32X4
  // A: 16x4 f32 (lane l15 = row M; half-wave selects K pair {0,1}/{2,3})
  const int koff = 2 * h;
  for (int k0 = 0; k0 < DIM; k0 += 4) {
    v2f a[2], b[4];
#pragma unroll
    for (int mi = 0; mi < 2; ++mi) a[mi] = *(const v2f*)(arow[mi] + k0 + koff);
#pragma unroll
    for (int ni = 0; ni < 4; ++ni) b[ni] = *(const v2f*)(brow[ni] + k0 + koff);
#pragma unroll
    for (int mi = 0; mi < 2; ++mi)
#pragma unroll
      for (int ni = 0; ni < 4; ++ni)
        acc[mi][ni] = __builtin_amdgcn_wmma_f32_16x16x4_f32(
            false, a[mi], false, b[ni], (short)0, acc[mi][ni], false, false);
  }
#else
  // Fallback: bf16 16x16x32 WMMA (probe-confirmed builtin)
  for (int k0 = 0; k0 < DIM; k0 += 32) {
    v16bf a[2], b[4];
#pragma unroll
    for (int mi = 0; mi < 2; ++mi) {
#pragma unroll
      for (int i = 0; i < 8; ++i) a[mi][i]     = (__bf16)arow[mi][k0 + 8 * h + i];
#pragma unroll
      for (int i = 0; i < 8; ++i) a[mi][8 + i] = (__bf16)arow[mi][k0 + 16 + 8 * h + i];
    }
#pragma unroll
    for (int ni = 0; ni < 4; ++ni) {
#pragma unroll
      for (int i = 0; i < 8; ++i) b[ni][i]     = (__bf16)brow[ni][k0 + 8 * h + i];
#pragma unroll
      for (int i = 0; i < 8; ++i) b[ni][8 + i] = (__bf16)brow[ni][k0 + 16 + 8 * h + i];
    }
#pragma unroll
    for (int mi = 0; mi < 2; ++mi)
#pragma unroll
      for (int ni = 0; ni < 4; ++ni)
        acc[mi][ni] = __builtin_amdgcn_wmma_f32_16x16x32_bf16(
            false, a[mi], false, b[ni], (short)0, acc[mi][ni], false, false);
  }
#endif

  // Epilogue: C/D layout — VGPR r, lanes 0-15: M=r, N=lane; lanes 16-31: M=r+8
  float lmax = 0.0f;
#pragma unroll
  for (int mi = 0; mi < 2; ++mi)
#pragma unroll
    for (int ni = 0; ni < 4; ++ni)
#pragma unroll
      for (int r = 0; r < 8; ++r) {
        int row = mbase + mi * 16 + r + (h << 3);
        int col = nbase + ni * 16 + l15;
        float dot = acc[mi][ni][r];
        float d = fmaxf(sq[row] + sq[col] - 2.0f * dot, 0.0f);
        Dm[(size_t)row * N_TOT + col] = d;
        lmax = fmaxf(lmax, d);
      }

  __shared__ float smax[256];
  smax[threadIdx.x] = lmax; __syncthreads();
  for (int off = 128; off > 0; off >>= 1) {
    if (threadIdx.x < off) smax[threadIdx.x] = fmaxf(smax[threadIdx.x], smax[threadIdx.x + off]);
    __syncthreads();
  }
  if (threadIdx.x == 0) atomicMax(maxbits, __float_as_int(smax[0]));  // valid: all values >= 0
}

// ---------------- median (histogram refinement) ----------------
__global__ void init_state_kernel(float* stateF, unsigned long long* rank, const int* maxbits) {
  float mx = __int_as_float(*maxbits);
  stateF[0] = 0.0f;
  stateF[1] = fmaxf(mx * 1.000001f + 1e-12f, 1e-6f);
  *rank = 8386559ULL;  // (N^2 - N)/2 - 1  (lower of the two central ranks)
}

__global__ __launch_bounds__(256) void hist_kernel(const float* __restrict__ Dm,
                                                   const float* __restrict__ stateF,
                                                   unsigned* __restrict__ hist) {
  __shared__ unsigned h[512];
  for (int i = threadIdx.x; i < 512; i += blockDim.x) h[i] = 0u;
  __syncthreads();
  float lo = stateF[0], hi = stateF[1];
  float scale = 512.0f / (hi - lo);
  int n = N_TOT * N_TOT;
  for (int idx = blockIdx.x * blockDim.x + threadIdx.x; idx < n; idx += gridDim.x * blockDim.x) {
    int i = idx >> 12, j = idx & (N_TOT - 1);
    if (i == j) continue;                    // off-diagonal only
    float d = Dm[idx];
    if (d >= lo && d < hi) {
      int b = (int)((d - lo) * scale);
      b = b < 0 ? 0 : (b > 511 ? 511 : b);
      atomicAdd(&h[b], 1u);
    }
  }
  __syncthreads();
  for (int i = threadIdx.x; i < 512; i += blockDim.x)
    if (h[i]) atomicAdd(&hist[i], h[i]);
}

__global__ void refine_kernel(float* stateF, unsigned long long* rank, unsigned* hist) {
  unsigned long long r = *rank, cum = 0;
  int b = 0;
  for (b = 0; b < 512; ++b) {
    unsigned c = hist[b];
    if (cum + c > r) break;
    cum += c;
  }
  if (b >= 512) { b = 511; cum -= hist[511]; }
  float lo = stateF[0], hi = stateF[1];
  float w = (hi - lo) * (1.0f / 512.0f);
  float nlo = lo + w * (float)b;
  float nhi = (b == 511) ? hi : lo + w * (float)(b + 1);
  if (!(nhi > nlo)) nhi = nlo + fmaxf(fabsf(nlo) * 1e-7f, 1e-30f);
  stateF[0] = nlo; stateF[1] = nhi; *rank = r - cum;
  for (int i = 0; i < 512; ++i) hist[i] = 0u;
}

// ---------------- _update_beta: per-pair stats ----------------
__global__ __launch_bounds__(128) void pair_stats_kernel(const float* __restrict__ S,
                                                         const float* __restrict__ Tt,
                                                         float* __restrict__ g) {
  __shared__ float red[6][128];
  int p = blockIdx.x, t = threadIdx.x;
  const float* v0 = S  + (size_t)(2 * p) * DIM;
  const float* v1 = v0 + DIM;
  const float* v2 = Tt + (size_t)(2 * p) * DIM;
  const float* v3 = v2 + DIM;
  float s01 = 0, s02 = 0, s03 = 0, s12 = 0, s13 = 0, s23 = 0;
  for (int j = t; j < DIM; j += 128) {
    float a = v0[j], b = v1[j], c = v2[j], d = v3[j], x;
    x = a - b; s01 += x * x;
    x = a - c; s02 += x * x;
    x = a - d; s03 += x * x;
    x = b - c; s12 += x * x;
    x = b - d; s13 += x * x;
    x = c - d; s23 += x * x;
  }
  red[0][t] = s01; red[1][t] = s02; red[2][t] = s03;
  red[3][t] = s12; red[4][t] = s13; red[5][t] = s23;
  __syncthreads();
  for (int off = 64; off > 0; off >>= 1) {
    if (t < off)
#pragma unroll
      for (int q = 0; q < 6; ++q) red[q][t] += red[q][t + off];
    __syncthreads();
  }
  if (t == 0) {
    float q[6];
#pragma unroll
    for (int i = 0; i < 6; ++i) q[i] = red[i][0];
    float srt[6];
#pragma unroll
    for (int i = 0; i < 6; ++i) srt[i] = q[i];
    for (int i = 1; i < 6; ++i) {          // insertion sort (6 elems)
      float key = srt[i]; int j = i - 1;
      while (j >= 0 && srt[j] > key) { srt[j + 1] = srt[j]; --j; }
      srt[j + 1] = key;
    }
    // median of 12 duplicated off-diag values = mean of middle two uniques
    float bw = 0.5f * (srt[2] + srt[3]) + 1e-6f;
    bw *= 0.25f;                            // / KERNEL_MUL^(KERNEL_NUM//2) = /4
    float inv = -1.0f / bw;
    float e01 = expf(q[0] * inv);           // sq(s0,s1)
    float e23 = expf(q[5] * inv);           // sq(t0,t1)
    float e02 = expf(q[1] * inv);           // sq(s0,t0)
    float e13 = expf(q[4] * inv);           // sq(s1,t1)
#pragma unroll
    for (int k = 0; k < KNUM; ++k) {
      g[p * KNUM + k] = e01 + e23 - e02 - e13;
      e01 = sqrtf(e01); e23 = sqrtf(e23); e02 = sqrtf(e02); e13 = sqrtf(e13);
    }
  }
}

// ---------------- Q matrix + 100 projected-gradient iterations ----------------
__global__ __launch_bounds__(128) void qmat_beta_kernel(const float* __restrict__ g,
                                                        const float* __restrict__ beta0,
                                                        float* __restrict__ betaOut) {
  __shared__ float gmS[KNUM];
  __shared__ float MS[KNUM * KNUM];
  int t = threadIdx.x;
  if (t < KNUM) gmS[t] = 0.0f;
  if (t < KNUM * KNUM) MS[t] = 0.0f;
  __syncthreads();
  float lg[KNUM] = {}, lM[KNUM * KNUM] = {};
  for (int p = t; p < PPAIR; p += 128) {
    float gv[KNUM];
#pragma unroll
    for (int k = 0; k < KNUM; ++k) gv[k] = g[p * KNUM + k];
#pragma unroll
    for (int a = 0; a < KNUM; ++a) {
      lg[a] += gv[a];
#pragma unroll
      for (int b = 0; b < KNUM; ++b) lM[a * KNUM + b] += gv[a] * gv[b];
    }
  }
#pragma unroll
  for (int a = 0; a < KNUM; ++a) atomicAdd(&gmS[a], lg[a]);
#pragma unroll
  for (int i = 0; i < KNUM * KNUM; ++i) atomicAdd(&MS[i], lM[i]);
  __syncthreads();

  if (t == 0) {
    const float invP = 1.0f / (float)PPAIR;
    float Q[KNUM * KNUM], gm[KNUM];
#pragma unroll
    for (int a = 0; a < KNUM; ++a) gm[a] = gmS[a] * invP;
#pragma unroll
    for (int a = 0; a < KNUM; ++a)
#pragma unroll
      for (int b = 0; b < KNUM; ++b)
        Q[a * KNUM + b] = MS[a * KNUM + b] * invP - gm[a] * gm[b] + ((a == b) ? EPSQ : 0.0f);

    float bv[KNUM];
#pragma unroll
    for (int a = 0; a < KNUM; ++a) bv[a] = beta0[a];

    for (int it = 0; it < 100; ++it) {
      float v[KNUM];
#pragma unroll
      for (int a = 0; a < KNUM; ++a) {
        float qa = 0.0f;
#pragma unroll
        for (int b = 0; b < KNUM; ++b) qa += Q[a * KNUM + b] * bv[b];
        v[a] = bv[a] - qa;
      }
      // simplex projection (exactly mirrors _project_simplex)
      float mu[KNUM];
#pragma unroll
      for (int a = 0; a < KNUM; ++a) mu[a] = v[a];
      for (int i = 1; i < KNUM; ++i) {     // sort descending
        float key = mu[i]; int j = i - 1;
        while (j >= 0 && mu[j] < key) { mu[j + 1] = mu[j]; --j; }
        mu[j + 1] = key;
      }
      float cs[KNUM], c = 0.0f;
#pragma unroll
      for (int i = 0; i < KNUM; ++i) { c += mu[i]; cs[i] = c; }
      int rho = 0; bool any = false;
#pragma unroll
      for (int i = 0; i < KNUM; ++i)
        if (mu[i] - (cs[i] - 1.0f) / (float)(i + 1) > 0.0f) { rho = i; any = true; }
      if (!any) rho = 0;
      float theta = (cs[rho] - 1.0f) / (float)(rho + 1);
      float s = 0.0f, pv[KNUM];
#pragma unroll
      for (int a = 0; a < KNUM; ++a) { pv[a] = fmaxf(v[a] - theta, 0.0f); s += pv[a]; }
      float invs = 1.0f / (s + 1e-12f);
#pragma unroll
      for (int a = 0; a < KNUM; ++a) bv[a] = pv[a] * invs;
    }
#pragma unroll
    for (int a = 0; a < KNUM; ++a) betaOut[a] = bv[a];
  }
}

// ---------------- final pass: kernels + block sums ----------------
__global__ __launch_bounds__(256) void loss_sums_kernel(const float* __restrict__ Dm,
                                                        const float* __restrict__ stateF,
                                                        const float* __restrict__ beta,
                                                        double* __restrict__ sums) {
  float med = 0.5f * (stateF[0] + stateF[1]) + 1e-6f;
  float bw = med * 0.25f;
  float inv = -1.0f / bw;
  float b0 = beta[0], b1 = beta[1], b2 = beta[2], b3 = beta[3], b4 = beta[4];
  double sxx = 0.0, syy = 0.0, sxy = 0.0;
  int n = N_TOT * N_TOT;
  for (int idx = blockIdx.x * blockDim.x + threadIdx.x; idx < n; idx += gridDim.x * blockDim.x) {
    int i = idx >> 12, j = idx & (N_TOT - 1);
    float d = Dm[idx];
    float e = expf(d * inv);                // bw * 2^k  ->  sqrt chain
    float jnt = b0 * e;
    e = sqrtf(e); jnt += b1 * e;
    e = sqrtf(e); jnt += b2 * e;
    e = sqrtf(e); jnt += b3 * e;
    e = sqrtf(e); jnt += b4 * e;
    if (i < NS && j < NS)        { if (i != j) sxx += (double)jnt; }
    else if (i >= NS && j >= NS) { if (i != j) syy += (double)jnt; }
    else if (i < NS)             { sxy += (double)jnt; }          // upper-right XY block only
  }
  __shared__ double red[3][256];
  int t = threadIdx.x;
  red[0][t] = sxx; red[1][t] = syy; red[2][t] = sxy;
  __syncthreads();
  for (int off = 128; off > 0; off >>= 1) {
    if (t < off) { red[0][t] += red[0][t + off]; red[1][t] += red[1][t + off]; red[2][t] += red[2][t + off]; }
    __syncthreads();
  }
  if (t == 0) {
    atomicAdd(&sums[0], red[0][0]);
    atomicAdd(&sums[1], red[1][0]);
    atomicAdd(&sums[2], red[2][0]);
  }
}

__global__ void finalize_kernel(const double* sums, float* out) {
  const double n = (double)NS, m = (double)NS;
  double loss = sums[0] / (n * (n - 1.0)) + sums[1] / (m * (m - 1.0)) - 2.0 * sums[2] / (n * m);
  out[0] = (float)(loss > 0.0 ? loss : 0.0);
}

// ---------------------------------------------------------------------------
extern "C" void kernel_launch(void* const* d_in, const int* in_sizes, int n_in,
                              void* d_out, int out_size, void* d_ws, size_t ws_size,
                              hipStream_t stream) {
  (void)in_sizes; (void)n_in; (void)out_size; (void)ws_size;
  const float* src   = (const float*)d_in[0];
  const float* tgt   = (const float*)d_in[1];
  const float* beta0 = (const float*)d_in[2];

  char* ws = (char*)d_ws;
  float*              Dm     = (float*)(ws + OFF_DMAT);
  float*              sq     = (float*)(ws + OFF_SQ);
  float*              Tot    = (float*)(ws + OFF_TOTAL);
  float*              g      = (float*)(ws + OFF_G);
  float*              betaN  = (float*)(ws + OFF_BETA);
  float*              stateF = (float*)(ws + OFF_STATE);
  unsigned long long* rank   = (unsigned long long*)(ws + OFF_STATE + 8);
  int*                maxb   = (int*)(ws + OFF_MAX);
  unsigned*           hist   = (unsigned*)(ws + OFF_HIST);
  double*             sums   = (double*)(ws + OFF_SUMS);

  zero_kernel<<<1, 512, 0, stream>>>(hist, sums, maxb);
  concat_kernel<<<2048, 256, 0, stream>>>(src, tgt, Tot);
  sqnorm_kernel<<<N_TOT, 128, 0, stream>>>(Tot, sq);
  gemm_dist_kernel<<<dim3(32, 32), 256, 0, stream>>>(Tot, sq, Dm, maxb);
  init_state_kernel<<<1, 1, 0, stream>>>(stateF, rank, maxb);
  for (int pass = 0; pass < 3; ++pass) {
    hist_kernel<<<2048, 256, 0, stream>>>(Dm, stateF, hist);
    refine_kernel<<<1, 1, 0, stream>>>(stateF, rank, hist);
  }
  pair_stats_kernel<<<PPAIR, 128, 0, stream>>>(src, tgt, g);
  qmat_beta_kernel<<<1, 128, 0, stream>>>(g, beta0, betaN);
  loss_sums_kernel<<<4096, 256, 0, stream>>>(Dm, stateF, betaN, sums);
  finalize_kernel<<<1, 1, 0, stream>>>(sums, (float*)d_out);
}
